// GraphConvolutionLayer_17403207483981
// MI455X (gfx1250) — matrix operand order repaired and checked by reading, not compiled, not run on hardware
//
#include <hip/hip_runtime.h>
#include <hip/hip_bf16.h>
#include <stddef.h>
#include <stdint.h>

// GCN layer for MI455X (gfx1250, wave32, WMMA).
// N = 8192 nodes, F_IN = F_OUT = 256.
//
// out[r,:] = d[r] * ( sum_k adj[r,k] * (d[k]*support[k,:]) + d[r]*support[r,:] ) + bias
// with d = rsqrt(rowsum(adj) + 1), support = input @ weight.
//
// adj (256MB fp32) > L2 (192MB): stream it from HBM exactly twice (degree pass +
// GEMM pass), convert to bf16 in-register, accumulate with v_wmma_f32_16x16x32_bf16.
// Pre-scaled support stored transposed in bf16 (4MB, L2-resident). The aggregate
// GEMM double-buffers its B panel in LDS via global_load_async_to_lds_b128
// (ASYNCcnt) and software-pipelines the fp32 A-frag loads.

#define NN   8192
#define FF   256
#define KB   32            // WMMA K per step (bf16)
#define LDS_ROW 96         // bytes per staged B row (64 data + 32 pad, 32B aligned)
#define PANEL   (FF * LDS_ROW)   // 24576 bytes per B panel

typedef __attribute__((ext_vector_type(16))) __bf16 v16bf;
typedef __attribute__((ext_vector_type(8)))  float  v8f;
typedef __attribute__((ext_vector_type(4)))  float  v4f;

// ---------------------------------------------------------------- degree pass
__global__ __launch_bounds__(256) void gcn_deg_kernel(
    const float* __restrict__ adj, float* __restrict__ dvec)
{
    const int row = blockIdx.x;
    const float* a = adj + (size_t)row * NN;
    float s = 0.0f;
    for (int k = threadIdx.x; k < NN; k += 256) s += a[k];
    #pragma unroll
    for (int off = 16; off > 0; off >>= 1) s += __shfl_down(s, off, 32);
    __shared__ float red[8];
    const int wave = threadIdx.x >> 5;
    if ((threadIdx.x & 31) == 0) red[wave] = s;
    __syncthreads();
    if (threadIdx.x == 0) {
        float t = red[0];
        #pragma unroll
        for (int i = 1; i < 8; ++i) t += red[i];
        dvec[row] = rsqrtf(t + 1.0f);   // +1 for the adj+I diagonal
    }
}

// -------------------------------------------- weight -> bf16, transposed [n][k]
__global__ __launch_bounds__(256) void gcn_wt_kernel(
    const float* __restrict__ w, __bf16* __restrict__ wt)
{
    const int k = blockIdx.x;      // 0..255
    const int n = threadIdx.x;     // 0..255
    wt[(size_t)n * FF + k] = (__bf16)w[(size_t)k * FF + n];
}

// ------------------------------------------------------------- WMMA helpers
__device__ inline void load_arows(const float* __restrict__ p,
                                  v4f& x0, v4f& x1, v4f& x2, v4f& x3)
{
    x0 = *(const v4f*)(p + 0);
    x1 = *(const v4f*)(p + 4);
    x2 = *(const v4f*)(p + 8);
    x3 = *(const v4f*)(p + 12);
}

__device__ inline v16bf cvt_afrag(v4f a0, v4f a1, v4f a2, v4f a3)
{
    // 16-bit A 16x32 layout used here: per lane, elements 0..15 -> K+0..15
    // (K already offset by 16 for lanes 16..31), matching B's contiguous order.
    v16bf af;
    #pragma unroll
    for (int i = 0; i < 4; ++i) {
        af[i]      = (__bf16)a0[i];
        af[4 + i]  = (__bf16)a1[i];
        af[8 + i]  = (__bf16)a2[i];
        af[12 + i] = (__bf16)a3[i];
    }
    return af;
}

// ------------------------------------------- CDNA5 async global->LDS staging
__device__ inline void async_b128(uint32_t lds_off, const void* gaddr)
{
    // GV mode: VDST = per-lane LDS byte address, VADDR = 64-bit global address.
    asm volatile("global_load_async_to_lds_b128 %0, %1, off"
                 :: "v"(lds_off), "v"(gaddr) : "memory");
}

__device__ inline void wait_async0()
{
    asm volatile("s_wait_asynccnt 0x0" ::: "memory");
}

// Stage one 256-row x 64-byte B panel (bf16, k-contiguous) into LDS buffer.
// 128 threads: 2 rows each, 4 async b128 per row (8 ASYNCcnt per wave).
__device__ inline void stage_panel_async(uint32_t buf_base,
                                         const __bf16* __restrict__ src,
                                         int k0, int tid)
{
    #pragma unroll
    for (int rr = 0; rr < 2; ++rr) {
        const int r = tid + rr * 128;
        const char* g = (const char*)(src + (size_t)r * NN + k0);
        const uint32_t l = buf_base + (uint32_t)r * LDS_ROW;
        #pragma unroll
        for (int i = 0; i < 4; ++i)
            async_b128(l + 16u * i, g + 16 * i);
    }
}

// ---------------------------------------------------------------- support GEMM
// support' = d[k] * (input @ weight), stored bf16 transposed: sbf[n][k]
__global__ __launch_bounds__(128) void gcn_support_kernel(
    const float*  __restrict__ x,     // [8192, 256]
    const __bf16* __restrict__ wt,    // [256 n][256 k]
    const float*  __restrict__ dvec,  // [8192]
    __bf16*       __restrict__ sbf)   // [256 n][8192 k]
{
    __shared__ __align__(32) unsigned char lds[PANEL];
    const int lane = threadIdx.x & 31;
    const int wave = threadIdx.x >> 5;
    const int lrow = lane & 15;
    const int lhi  = lane >> 4;                 // 0 or 1
    const int row0 = blockIdx.x * 64 + wave * 16;
    const int arow = row0 + lrow;

    v8f acc[16];
    const v8f zero = {0.f,0.f,0.f,0.f,0.f,0.f,0.f,0.f};
    #pragma unroll
    for (int j = 0; j < 16; ++j) acc[j] = zero;

    for (int k0 = 0; k0 < FF; k0 += KB) {
        __syncthreads();
        for (int r = threadIdx.x; r < FF; r += 128) {           // stage B panel
            const uint4* src = (const uint4*)(wt + (size_t)r * FF + k0);
            uint4* dst = (uint4*)(lds + r * LDS_ROW);
            dst[0] = src[0]; dst[1] = src[1]; dst[2] = src[2]; dst[3] = src[3];
        }
        __syncthreads();
        v4f a0, a1, a2, a3;
        load_arows(x + (size_t)arow * FF + k0 + lhi * 16, a0, a1, a2, a3);
        const v16bf af = cvt_afrag(a0, a1, a2, a3);
        #pragma unroll
        for (int j = 0; j < 16; ++j) {
            const v16bf bf = *(const v16bf*)(lds + (j * 16 + lrow) * LDS_ROW + lhi * 32);
            acc[j] = __builtin_amdgcn_wmma_f32_16x16x32_bf16(
                false, af, false, bf, (short)0, acc[j], false, false);
        }
    }

    // epilogue: sbf[n][r] = bf16(d[r] * acc)  (C layout: M = v + 8*lhi, N = j*16+lrow)
    #pragma unroll
    for (int j = 0; j < 16; ++j) {
        const int n = j * 16 + lrow;
        #pragma unroll
        for (int v = 0; v < 8; ++v) {
            const int r = row0 + v + lhi * 8;
            sbf[(size_t)n * NN + r] = (__bf16)(dvec[r] * acc[j][v]);
        }
    }
}

// -------------------------------------------------------------- aggregate GEMM
// out[r,n] = d[r] * ( sum_k adj[r,k]*sbf[n][k] + sbf[n][r] ) + bias[n]
// Double-buffered async B staging + software-pipelined A-frag loads.
__global__ __launch_bounds__(128) void gcn_agg_kernel(
    const float*  __restrict__ adj,   // [8192, 8192] fp32, streamed from HBM
    const __bf16* __restrict__ sbf,   // [256 n][8192 k] bf16, L2-resident
    const float*  __restrict__ dvec,
    const float*  __restrict__ bias,
    float*        __restrict__ out)   // [8192, 256]
{
    __shared__ __align__(32) unsigned char lds[2 * PANEL];
    const int tid  = threadIdx.x;
    const int lane = tid & 31;
    const int wave = tid >> 5;
    const int lrow = lane & 15;
    const int lhi  = lane >> 4;
    const int row0 = blockIdx.x * 64 + wave * 16;
    const float* arow_base = adj + (size_t)(row0 + lrow) * NN + lhi * 16;

    v8f acc[16];
    const v8f zero = {0.f,0.f,0.f,0.f,0.f,0.f,0.f,0.f};
    #pragma unroll
    for (int j = 0; j < 16; ++j) acc[j] = zero;

    // prologue: stage panel 0 asynchronously, prefetch A-frag 0
    stage_panel_async(0u, sbf, 0, tid);
    v4f a0, a1, a2, a3;
    load_arows(arow_base, a0, a1, a2, a3);
    wait_async0();
    __syncthreads();

    const int nIter = NN / KB;                   // 256
    for (int it = 0; it < nIter; ++it) {
        const uint32_t cur = (it & 1) ? (uint32_t)PANEL : 0u;
        const uint32_t nxt = (it & 1) ? 0u : (uint32_t)PANEL;
        const bool more = (it + 1) < nIter;
        const int k1 = (it + 1) * KB;

        if (more) stage_panel_async(nxt, sbf, k1, tid);   // overlap with WMMAs

        v4f b0, b1, b2, b3;                               // prefetch next A-frag
        if (more) load_arows(arow_base + k1, b0, b1, b2, b3);

        const v16bf af = cvt_afrag(a0, a1, a2, a3);
        #pragma unroll
        for (int j = 0; j < 16; ++j) {
            const v16bf bf = *(const v16bf*)(lds + cur + (j * 16 + lrow) * LDS_ROW + lhi * 32);
            acc[j] = __builtin_amdgcn_wmma_f32_16x16x32_bf16(
                false, af, false, bf, (short)0, acc[j], false, false);
        }
        a0 = b0; a1 = b1; a2 = b2; a3 = b3;

        if (more) {
            wait_async0();      // this wave's writes to nxt complete...
            __syncthreads();    // ...and everyone else's too
        }
    }

    #pragma unroll
    for (int j = 0; j < 16; ++j) {
        const int n = j * 16 + lrow;
        const float bn = bias[n];
        #pragma unroll
        for (int v = 0; v < 8; ++v) {
            const int r = row0 + v + lhi * 8;
            const float ident = (float)sbf[(size_t)n * NN + r];   // +I term
            out[(size_t)r * FF + n] = dvec[r] * (acc[j][v] + ident) + bn;
        }
    }
}

// ------------------------------------------------------------------- launcher
extern "C" void kernel_launch(void* const* d_in, const int* in_sizes, int n_in,
                              void* d_out, int out_size, void* d_ws, size_t ws_size,
                              hipStream_t stream) {
    const float* input  = (const float*)d_in[0];   // [8192, 256]
    const float* adj    = (const float*)d_in[1];   // [8192, 8192]
    const float* weight = (const float*)d_in[2];   // [256, 256]
    const float* bias   = (const float*)d_in[3];   // [256]
    float* out = (float*)d_out;

    // workspace: d (32KB) | Wt bf16 (128KB) | Sbf bf16 (4MB)  ~= 4.4MB total
    char* ws = (char*)d_ws;
    float*  dvec = (float*)ws;
    __bf16* wt   = (__bf16*)(ws + 32768);
    __bf16* sbf  = (__bf16*)(ws + 32768 + 131072);

    gcn_deg_kernel<<<NN, 256, 0, stream>>>(adj, dvec);
    gcn_wt_kernel<<<FF, 256, 0, stream>>>(weight, wt);
    gcn_support_kernel<<<NN / 64, 128, 0, stream>>>(input, wt, dvec, sbf);
    gcn_agg_kernel<<<NN / 64, 128, 0, stream>>>(adj, sbf, dvec, bias, out);
}
